// LinearTemporalDiffusionTransformerDecoderLayer_83236466197123
// MI455X (gfx1250) — compile-verified, matrix operand order checked by reading, not run on hardware
//
#include <hip/hip_runtime.h>
#include <math.h>

#define B_   16
#define T_   8192
#define N_   77
#define D_   256
#define DT_  512
#define TE_  512
#define FD_  1024
#define H_   8
#define HD_  32
#define BT_  (B_*T_)
#define EPS_ 1e-5f

typedef __bf16 bf16;
typedef __bf16 bf16x16 __attribute__((ext_vector_type(16)));
typedef __bf16 bf16x8  __attribute__((ext_vector_type(8)));
typedef __bf16 bf16x4  __attribute__((ext_vector_type(4)));
typedef float  f32x8   __attribute__((ext_vector_type(8)));
typedef float  f32x4   __attribute__((ext_vector_type(4)));

__device__ __forceinline__ f32x8 zero8() {
  f32x8 z = {0.f,0.f,0.f,0.f,0.f,0.f,0.f,0.f};
  return z;
}
__device__ __forceinline__ float silu_f(float x)  { return x / (1.f + __expf(-x)); }
__device__ __forceinline__ float gelu_f(float x)  { return 0.5f*x*(1.f+erff(x*0.70710678118654752f)); }

// ---- Async global->LDS copy of a 32x256 fp32 tile (CDNA5 async path) ----
__device__ __forceinline__ void async_copy_tile_f32(const float* gbase, float* lbase) {
  unsigned loff = (unsigned)(size_t)lbase;     // LDS aperture: low 32 bits = LDS offset
  int tid = threadIdx.x;
#pragma unroll
  for (int i = 0; i < 8; ++i) {
    unsigned goff = (unsigned)((i*1024 + tid*4) * 4);
    unsigned lo = loff + goff;
    asm volatile("global_load_async_to_lds_b128 %0, %1, %2"
                 :: "v"(lo), "v"(goff), "s"(gbase) : "memory");
  }
  asm volatile("s_wait_asynccnt 0x0" ::: "memory");
}

// ---- WMMA fragment helpers (wave32 layouts per CDNA5 ISA 7.12.2) ----
__device__ __forceinline__ bf16x16 frag_kmaj(const bf16* p, long ld) {
  int lane = threadIdx.x & 31;
  int sel = lane & 15, half = lane >> 4;
  const bf16* r0 = p + (size_t)sel*ld + half*8;
  bf16x8 lo = *(const bf16x8*)(r0);
  bf16x8 hi = *(const bf16x8*)(r0 + 16);
  return __builtin_shufflevector(lo, hi, 0,1,2,3,4,5,6,7,8,9,10,11,12,13,14,15);
}
__device__ __forceinline__ bf16x16 frag_kmaj_exp(const bf16* p, long ld, float cm, float ci) {
  int lane = threadIdx.x & 31;
  int sel = lane & 15, half = lane >> 4;
  const bf16* r0 = p + (size_t)sel*ld + half*8;
  bf16x8 lo = *(const bf16x8*)(r0);
  bf16x8 hi = *(const bf16x8*)(r0 + 16);
  bf16x16 f;
#pragma unroll
  for (int i = 0; i < 8; ++i) {
    f[i]   = (bf16)(__expf((float)lo[i] - cm)*ci);
    f[8+i] = (bf16)(__expf((float)hi[i] - cm)*ci);
  }
  return f;
}
__device__ __forceinline__ bf16x16 frag_attB(const float* base) {
  int lane = threadIdx.x & 31;
  int n = lane & 15, half = lane >> 4;
  const float* r0 = base + n*32 + half*8;
  f32x4 a0 = *(const f32x4*)(r0);
  f32x4 a1 = *(const f32x4*)(r0 + 4);
  f32x4 b0 = *(const f32x4*)(r0 + 16);
  f32x4 b1 = *(const f32x4*)(r0 + 20);
  bf16x16 f;
#pragma unroll
  for (int i = 0; i < 4; ++i) {
    f[i] = (bf16)a0[i]; f[4+i] = (bf16)a1[i];
    f[8+i] = (bf16)b0[i]; f[12+i] = (bf16)b1[i];
  }
  return f;
}
__device__ __forceinline__ f32x8 wmma_bf16(bf16x16 a, bf16x16 b, f32x8 c) {
  return __builtin_amdgcn_wmma_f32_16x16x32_bf16(false, a, false, b, (short)0, c, false, false);
}
__device__ __forceinline__ void store_c_lds(float* p, int ld, f32x8 c) {
  int lane = threadIdx.x & 31;
  int n = lane & 15, half = lane >> 4;
#pragma unroll
  for (int r = 0; r < 8; ++r)
    p[(r + half*8)*ld + n] = c[r];
}
// Stage a 32x256 fp32 weight slab (row stride ldw) transposed into swt[256][40] bf16.
// 4x4 sub-blocks: 4x b128 global loads up front, then 4x b64 LDS stores (k-quads).
__device__ __forceinline__ void stage_w_t(const float* W, int ldw, bf16* swt) {
  int tid = threadIdx.x;
  const f32x4* Wr = (const f32x4*)W;
  f32x4 v[8];
#pragma unroll
  for (int u = 0; u < 2; ++u) {
    int idx = u*256 + tid;
    int rq = idx >> 6, cq = idx & 63;
#pragma unroll
    for (int j = 0; j < 4; ++j)
      v[u*4+j] = Wr[(size_t)(rq*4 + j)*(ldw >> 2) + cq];
  }
#pragma unroll
  for (int u = 0; u < 2; ++u) {
    int idx = u*256 + tid;
    int rq = idx >> 6, cq = idx & 63;
#pragma unroll
    for (int i = 0; i < 4; ++i) {
      bf16x4 pk;
#pragma unroll
      for (int j = 0; j < 4; ++j) pk[j] = (bf16)v[u*4+j][i];
      *(bf16x4*)(swt + (cq*4 + i)*40 + rq*4) = pk;
    }
  }
}

// ---- 32-row LayerNorm into bf16 tile (256 threads: 8 threads/row) ----
__device__ __forceinline__ void ln_rows_to_bf16(const float* src, bf16* dst, int ldd,
                                                const float* g, const float* bb) {
  __shared__ float prow[32][8];
  int tid = threadIdx.x;
  int r = tid >> 3, sub = tid & 7;
  float s = 0.f;
  for (int c = sub*32; c < sub*32+32; ++c) s += src[r*256 + c];
  prow[r][sub] = s;  __syncthreads();
  float mean = 0.f;
#pragma unroll
  for (int i = 0; i < 8; ++i) mean += prow[r][i];
  mean *= (1.f/256.f);
  __syncthreads();
  float vs = 0.f;
  for (int c = sub*32; c < sub*32+32; ++c) { float d = src[r*256+c]-mean; vs += d*d; }
  prow[r][sub] = vs;  __syncthreads();
  float var = 0.f;
#pragma unroll
  for (int i = 0; i < 8; ++i) var += prow[r][i];
  float rstd = rsqrtf(var*(1.f/256.f) + EPS_);
  for (int c = sub*32; c < sub*32+32; c += 4) {
    bf16x4 pk;
#pragma unroll
    for (int i = 0; i < 4; ++i)
      pk[i] = (bf16)((src[r*256+c+i]-mean)*rstd*g[c+i] + bb[c+i]);
    *(bf16x4*)(dst + r*ldd + c) = pk;
  }
  __syncthreads();
}

// ---- Stylization block: h -> LN -> FiLM -> SiLU -> @Wo + bo + residual ----
__device__ void styl_block(const float* sy, bf16* sa, bf16* swt0, bf16* swt1,
                           const float* filmS, int bidx,
                           const float* g, const float* bln,
                           const float* Wo, const float* bo,
                           const float* resptr, float* outptr, long rowbase) {
  __shared__ float prow[32][8];
  int tid = threadIdx.x, wave = tid >> 5, lane = tid & 31, half = lane >> 4, nn = lane & 15;
  __syncthreads();
  int r = tid >> 3, sub = tid & 7;
  float s = 0.f;
  for (int c = sub*32; c < sub*32+32; ++c) s += sy[r*256 + c];
  prow[r][sub] = s;  __syncthreads();
  float mean = 0.f;
#pragma unroll
  for (int i = 0; i < 8; ++i) mean += prow[r][i];
  mean *= (1.f/256.f);
  __syncthreads();
  float vs = 0.f;
  for (int c = sub*32; c < sub*32+32; ++c) { float d = sy[r*256+c]-mean; vs += d*d; }
  prow[r][sub] = vs;  __syncthreads();
  float var = 0.f;
#pragma unroll
  for (int i = 0; i < 8; ++i) var += prow[r][i];
  float rstd = rsqrtf(var*(1.f/256.f) + EPS_);
  for (int c = sub*32; c < sub*32+32; c += 4) {
    bf16x4 pk;
#pragma unroll
    for (int i = 0; i < 4; ++i) {
      float hn = (sy[r*256+c+i]-mean)*rstd*g[c+i] + bln[c+i];
      hn = hn*(1.f + filmS[bidx*512 + c+i]) + filmS[bidx*512 + 256 + c+i];
      pk[i] = (bf16)silu_f(hn);
    }
    *(bf16x4*)(sa + r*264 + c) = pk;
  }
  __syncthreads();
  f32x8 acc[4];
#pragma unroll
  for (int t = 0; t < 4; ++t) acc[t] = zero8();
  stage_w_t(Wo, D_, swt0);
  __syncthreads();
  for (int kk = 0; kk < 8; ++kk) {
    bf16* cur = (kk & 1) ? swt1 : swt0;
    bf16* nxt = (kk & 1) ? swt0 : swt1;
    if (kk < 6) __builtin_prefetch(Wo + (size_t)(kk+2)*32*D_, 0, 1);
    if (kk < 7) stage_w_t(Wo + (size_t)(kk+1)*32*D_, D_, nxt);
#pragma unroll
    for (int t = 0; t < 4; ++t) {
      int tile = wave*4 + t, mt = tile >> 4, nt = tile & 15;
      acc[t] = wmma_bf16(frag_kmaj(sa + mt*16*264 + kk*32, 264),
                         frag_kmaj(cur + nt*16*40, 40), acc[t]);
    }
    __syncthreads();
  }
#pragma unroll
  for (int t = 0; t < 4; ++t) {
    int tile = wave*4 + t, mt = tile >> 4, nt = tile & 15;
#pragma unroll
    for (int r2 = 0; r2 < 8; ++r2) {
      long grow = rowbase + mt*16 + r2 + half*8;
      int cc = nt*16 + nn;
      outptr[grow*D_ + cc] = acc[t][r2] + bo[cc] + resptr[grow*D_ + cc];
    }
  }
}

// ============ Kernel 1: FiLM params for the three styl blocks ============
__global__ void film_kernel(const float* emb,
                            const float* We0, const float* be0,
                            const float* We1, const float* be1,
                            const float* We2, const float* be2,
                            float* film) {
  int b = blockIdx.x;
  __shared__ float se[TE_];
  for (int i = threadIdx.x; i < TE_; i += blockDim.x) se[i] = silu_f(emb[b*TE_ + i]);
  __syncthreads();
  const float* We[3] = {We0, We1, We2};
  const float* be[3] = {be0, be1, be2};
  for (int s = 0; s < 3; ++s)
    for (int j = threadIdx.x; j < 2*D_; j += blockDim.x) {
      float acc = be[s][j];
      for (int i = 0; i < TE_; ++i) acc += se[i]*We[s][i*(2*D_) + j];
      film[(s*B_ + b)*(2*D_) + j] = acc;
    }
}

// ============ Kernel 2: CA context state; stores TRANSPOSED attT[e][d] ============
__global__ void ca_ctx_kernel(const float* xf, const float* tg, const float* tb,
                              const float* Wk, const float* bk,
                              const float* Wv, const float* bv,
                              float* attT) {
  int b = blockIdx.x / H_, h = blockIdx.x % H_;
  __shared__ float row[DT_];
  __shared__ float kcol[N_][HD_];
  __shared__ float vcol[N_][HD_];
  __shared__ float red[256];
  int tid = threadIdx.x;
  for (int n = 0; n < N_; ++n) {
    float v0 = xf[(b*N_ + n)*DT_ + tid];
    float v1 = xf[(b*N_ + n)*DT_ + tid + 256];
    red[tid] = v0 + v1; __syncthreads();
    for (int s = 128; s > 0; s >>= 1) { if (tid < s) red[tid] += red[tid+s]; __syncthreads(); }
    float mean = red[0]*(1.f/DT_); __syncthreads();
    float d0 = v0-mean, d1 = v1-mean;
    red[tid] = d0*d0 + d1*d1; __syncthreads();
    for (int s = 128; s > 0; s >>= 1) { if (tid < s) red[tid] += red[tid+s]; __syncthreads(); }
    float rstd = rsqrtf(red[0]*(1.f/DT_) + EPS_); __syncthreads();
    row[tid]     = d0*rstd*tg[tid] + tb[tid];
    row[tid+256] = d1*rstd*tg[tid+256] + tb[tid+256];
    __syncthreads();
    int col = tid & 63, part = tid >> 6, c = col & 31;
    const float* W = (col < 32) ? Wk : Wv;
    float acc = 0.f;
    for (int i = part*128; i < part*128 + 128; ++i) acc += row[i]*W[i*D_ + h*HD_ + c];
    red[tid] = acc; __syncthreads();
    if (part == 0) {
      float sm = red[col] + red[col+64] + red[col+128] + red[col+192];
      if (col < 32) kcol[n][c] = sm + bk[h*HD_ + c];
      else          vcol[n][c] = sm + bv[h*HD_ + c];
    }
    __syncthreads();
  }
  if (tid < HD_) {
    float mx = -1e30f;
    for (int n = 0; n < N_; ++n) mx = fmaxf(mx, kcol[n][tid]);
    float sum = 0.f;
    for (int n = 0; n < N_; ++n) { float e = __expf(kcol[n][tid]-mx); kcol[n][tid] = e; sum += e; }
    float inv = 1.f/sum;
    for (int n = 0; n < N_; ++n) kcol[n][tid] *= inv;
  }
  __syncthreads();
  for (int o = tid; o < HD_*HD_; o += 256) {
    int d = o >> 5, e = o & 31;
    float acc = 0.f;
    for (int n = 0; n < N_; ++n) acc += kcol[n][d]*vcol[n][e];
    attT[((b*H_ + h)*HD_ + e)*HD_ + d] = acc;   // transposed: d contiguous
  }
}

// ============ Kernel 3: SA fused LN + q/k/v WMMA GEMMs; k/v spilled TRANSPOSED ============
__global__ void sa_qkv_kernel(const float* x, const int* mask,
                              const float* ln_g, const float* ln_b,
                              const float* Wq, const float* bq,
                              const float* Wk, const float* bk,
                              const float* Wv, const float* bv,
                              bf16* qout, bf16* kT, bf16* vT) {
  extern __shared__ char smem[];
  float* sx   = (float*)smem;                     // [32][256]
  bf16*  sa   = (bf16*)(smem + 32768);            // [32][264]
  bf16*  swt0 = (bf16*)(smem + 49664);            // [256][40]
  bf16*  swt1 = (bf16*)(smem + 70144);            // [256][40]
  float* so   = (float*)(smem + 90624);           // [32][256]
  __shared__ int smask[32];
  int tid = threadIdx.x, wave = tid >> 5;
  long rowbase = (long)blockIdx.x * 32;
  int b = (int)(rowbase / T_);
  int tbase = (int)(rowbase % T_);
  async_copy_tile_f32(x + rowbase*D_, sx);
  if (tid < 32) smask[tid] = mask[b*T_ + tbase + tid];
  __syncthreads();
  ln_rows_to_bf16(sx, sa, 264, ln_g, ln_b);

  const float* Ws[3] = {Wq, Wk, Wv};
  const float* bs[3] = {bq, bk, bv};
  for (int mat = 0; mat < 3; ++mat) {
    f32x8 acc[4];
#pragma unroll
    for (int t = 0; t < 4; ++t) acc[t] = zero8();
    stage_w_t(Ws[mat], D_, swt0);
    __syncthreads();
    for (int kk = 0; kk < 8; ++kk) {
      bf16* cur = (kk & 1) ? swt1 : swt0;
      bf16* nxt = (kk & 1) ? swt0 : swt1;
      if (kk < 6) __builtin_prefetch(Ws[mat] + (size_t)(kk+2)*32*D_, 0, 1);
      if (kk < 7) stage_w_t(Ws[mat] + (size_t)(kk+1)*32*D_, D_, nxt);
#pragma unroll
      for (int t = 0; t < 4; ++t) {
        int tile = wave*4 + t, mt = tile >> 4, nt = tile & 15;
        acc[t] = wmma_bf16(frag_kmaj(sa + mt*16*264 + kk*32, 264),
                           frag_kmaj(cur + nt*16*40, 40), acc[t]);
      }
      __syncthreads();
    }
#pragma unroll
    for (int t = 0; t < 4; ++t) {
      int tile = wave*4 + t, mt = tile >> 4, nt = tile & 15;
      store_c_lds(so + mt*16*256 + nt*16, 256, acc[t]);
    }
    __syncthreads();
    if (mat == 0) {                       // q: bias + softmax over head dim
      int rr = tid >> 3, hh = tid & 7;
      float vals[32]; float mx = -1e30f;
#pragma unroll
      for (int e = 0; e < 32; ++e) {
        float v = so[rr*256 + hh*32 + e] + bq[hh*32 + e];
        vals[e] = v; mx = fmaxf(mx, v);
      }
      float sum = 0.f;
#pragma unroll
      for (int e = 0; e < 32; ++e) { float ev = __expf(vals[e]-mx); vals[e] = ev; sum += ev; }
      float inv = 1.f/sum;
#pragma unroll
      for (int v8 = 0; v8 < 4; ++v8) {
        bf16x8 pk;
#pragma unroll
        for (int i = 0; i < 8; ++i) pk[i] = (bf16)(vals[v8*8+i]*inv);
        *(bf16x8*)(qout + (rowbase + rr)*D_ + hh*32 + v8*8) = pk;
      }
    } else {                              // k/v: bias + mask, store transposed [c][t] as b128
      bf16* dst = (mat == 1) ? kT : vT;
#pragma unroll
      for (int u = 0; u < 4; ++u) {
        int idx = u*256 + tid;            // 1024 items: 256 cols x 4 t-octets
        int cc = idx >> 2, to = (idx & 3)*8;
        bf16x8 pk;
#pragma unroll
        for (int i = 0; i < 8; ++i) {
          int rr = to + i;
          float v = so[rr*256 + cc] + bs[mat][cc];
          int mk = smask[rr];
          v = (mat == 1) ? (mk ? -1e30f : v) : (mk ? 0.f : v);
          pk[i] = (bf16)v;
        }
        *(bf16x8*)(dst + ((size_t)(b*D_ + cc))*T_ + tbase + to) = pk;
      }
    }
    __syncthreads();
  }
}

// ============ Kernel 4: per-column (b,d) max & 1/sumexp over T (contiguous rows) ============
__global__ void sa_colstat_kernel(const bf16* kT, float* colmax, float* colinv) {
  int col = blockIdx.x;                         // b*256 + c
  const bf16* row = kT + (size_t)col * T_;
  __shared__ float red[256];
  float mx = -1e30f;
  for (int t = threadIdx.x*8; t < T_; t += 256*8) {
    bf16x8 v = *(const bf16x8*)(row + t);
#pragma unroll
    for (int i = 0; i < 8; ++i) mx = fmaxf(mx, (float)v[i]);
  }
  red[threadIdx.x] = mx; __syncthreads();
  for (int s = 128; s > 0; s >>= 1) { if (threadIdx.x < s) red[threadIdx.x] = fmaxf(red[threadIdx.x], red[threadIdx.x+s]); __syncthreads(); }
  mx = red[0]; __syncthreads();
  float sum = 0.f;
  for (int t = threadIdx.x*8; t < T_; t += 256*8) {
    bf16x8 v = *(const bf16x8*)(row + t);
#pragma unroll
    for (int i = 0; i < 8; ++i) sum += __expf((float)v[i] - mx);
  }
  red[threadIdx.x] = sum; __syncthreads();
  for (int s = 128; s > 0; s >>= 1) { if (threadIdx.x < s) red[threadIdx.x] += red[threadIdx.x+s]; __syncthreads(); }
  if (threadIdx.x == 0) { colmax[col] = mx; colinv[col] = 1.f/red[0]; }
}

// ============ Kernel 5: att[b,h] = K~^T V via WMMA over T (all-vector fragments) ============
__global__ void sa_att_kernel(const bf16* kT, const bf16* vT,
                              const float* colmax, const float* colinv,
                              float* att) {
  int b = blockIdx.x >> 3, h = blockIdx.x & 7;
  int tid = threadIdx.x, lane = tid & 31, wave = tid >> 5, half = lane >> 4;
  __shared__ float sred[8][32][32];
  f32x8 acc[2][2];
#pragma unroll
  for (int dt = 0; dt < 2; ++dt)
#pragma unroll
    for (int et = 0; et < 2; ++et) acc[dt][et] = zero8();
  float cm[2], ci[2];
#pragma unroll
  for (int dt = 0; dt < 2; ++dt) {
    int d = (lane & 15) + dt*16;
    cm[dt] = colmax[b*D_ + h*HD_ + d];
    ci[dt] = colinv[b*D_ + h*HD_ + d];
  }
  const bf16* kbase = kT + (size_t)(b*D_ + h*HD_)*T_;
  const bf16* vbase = vT + (size_t)(b*D_ + h*HD_)*T_;
  for (int chunk = 0; chunk < 32; ++chunk) {
    int t0 = wave*1024 + chunk*32;
    bf16x16 afr[2], bfr[2];
#pragma unroll
    for (int dt = 0; dt < 2; ++dt)
      afr[dt] = frag_kmaj_exp(kbase + (size_t)dt*16*T_ + t0, T_, cm[dt], ci[dt]);
#pragma unroll
    for (int et = 0; et < 2; ++et)
      bfr[et] = frag_kmaj(vbase + (size_t)et*16*T_ + t0, T_);
#pragma unroll
    for (int dt = 0; dt < 2; ++dt)
#pragma unroll
      for (int et = 0; et < 2; ++et)
        acc[dt][et] = wmma_bf16(afr[dt], bfr[et], acc[dt][et]);
  }
#pragma unroll
  for (int dt = 0; dt < 2; ++dt)
#pragma unroll
    for (int et = 0; et < 2; ++et)
#pragma unroll
      for (int r2 = 0; r2 < 8; ++r2)
        sred[wave][r2 + half*8 + dt*16][(lane & 15) + et*16] = acc[dt][et][r2];
  __syncthreads();
  for (int o = tid; o < 32*32; o += 256) {
    int d = o >> 5, e = o & 31;
    float s = 0.f;
#pragma unroll
    for (int w = 0; w < 8; ++w) s += sred[w][d][e];
    att[((b*H_ + h)*HD_ + e)*HD_ + d] = s;      // [e][d]: d contiguous for y-GEMM
  }
}

// ============ Kernel 6: SA y = q@att (per-head WMMA) + styl + residual ============
__global__ void sa_apply_kernel(const bf16* qbuf, const float* att,
                                const float* filmS, const float* sg, const float* sb,
                                const float* Wo, const float* bo,
                                const float* xres, float* out) {
  extern __shared__ char smem[];
  float* sy   = (float*)smem;                     // [32][256]
  bf16*  sa   = (bf16*)(smem + 32768);
  bf16*  swt0 = (bf16*)(smem + 49664);
  bf16*  swt1 = (bf16*)(smem + 70144);
  int tid = threadIdx.x, lane = tid & 31, wave = tid >> 5, half = lane >> 4;
  long rowbase = (long)blockIdx.x * 32;
  int bidx = (int)(rowbase / T_);
#pragma unroll
  for (int t = 0; t < 4; ++t) {
    int idx = wave*4 + t, mt = idx >> 4, rest = idx & 15, h = rest >> 1, et = rest & 1;
    bf16x16 af = frag_kmaj(qbuf + (rowbase + mt*16)*D_ + h*HD_, D_);
    bf16x16 bfv = frag_attB(att + ((size_t)(bidx*H_ + h)*HD_ + et*16)*HD_);
    f32x8 c = zero8();
    c = wmma_bf16(af, bfv, c);
    int n = lane & 15;
#pragma unroll
    for (int r2 = 0; r2 < 8; ++r2)
      sy[(mt*16 + r2 + half*8)*256 + h*HD_ + et*16 + n] = c[r2];
  }
  styl_block(sy, sa, swt0, swt1, filmS, bidx, sg, sb, Wo, bo, xres, out, rowbase);
}

// ============ Kernel 7: CA q proj + softmax + y (per-head WMMA) + styl ============
__global__ void ca_apply_kernel(const float* xin,
                                const float* ln_g, const float* ln_b,
                                const float* Wq, const float* bq,
                                const float* attT, const float* filmS,
                                const float* sg, const float* sb,
                                const float* Wo, const float* bo, float* out) {
  extern __shared__ char smem[];
  float* sy   = (float*)smem;
  bf16*  sa   = (bf16*)(smem + 32768);
  bf16*  swt0 = (bf16*)(smem + 49664);
  bf16*  swt1 = (bf16*)(smem + 70144);
  int tid = threadIdx.x, lane = tid & 31, wave = tid >> 5, half = lane >> 4;
  long rowbase = (long)blockIdx.x * 32;
  int bidx = (int)(rowbase / T_);
  async_copy_tile_f32(xin + rowbase*D_, sy);
  __syncthreads();
  ln_rows_to_bf16(sy, sa, 264, ln_g, ln_b);
  f32x8 acc[4];
#pragma unroll
  for (int t = 0; t < 4; ++t) acc[t] = zero8();
  stage_w_t(Wq, D_, swt0);
  __syncthreads();
  for (int kk = 0; kk < 8; ++kk) {
    bf16* cur = (kk & 1) ? swt1 : swt0;
    bf16* nxt = (kk & 1) ? swt0 : swt1;
    if (kk < 6) __builtin_prefetch(Wq + (size_t)(kk+2)*32*D_, 0, 1);
    if (kk < 7) stage_w_t(Wq + (size_t)(kk+1)*32*D_, D_, nxt);
#pragma unroll
    for (int t = 0; t < 4; ++t) {
      int tile = wave*4 + t, mt = tile >> 4, nt = tile & 15;
      acc[t] = wmma_bf16(frag_kmaj(sa + mt*16*264 + kk*32, 264),
                         frag_kmaj(cur + nt*16*40, 40), acc[t]);
    }
    __syncthreads();
  }
#pragma unroll
  for (int t = 0; t < 4; ++t) {
    int tile = wave*4 + t, mt = tile >> 4, nt = tile & 15;
    store_c_lds(sy + mt*16*256 + nt*16, 256, acc[t]);
  }
  __syncthreads();
  {
    int rr = tid >> 3, hh = tid & 7;
    float vals[32]; float mx = -1e30f;
#pragma unroll
    for (int e = 0; e < 32; ++e) {
      float v = sy[rr*256 + hh*32 + e] + bq[hh*32 + e];
      vals[e] = v; mx = fmaxf(mx, v);
    }
    float sum = 0.f;
#pragma unroll
    for (int e = 0; e < 32; ++e) { float ev = __expf(vals[e]-mx); vals[e] = ev; sum += ev; }
    float inv = 1.f/sum;
#pragma unroll
    for (int v4 = 0; v4 < 8; ++v4) {
      bf16x4 pk;
#pragma unroll
      for (int i = 0; i < 4; ++i) pk[i] = (bf16)(vals[v4*4+i]*inv);
      *(bf16x4*)(sa + rr*264 + hh*32 + v4*4) = pk;
    }
  }
  __syncthreads();
#pragma unroll
  for (int t = 0; t < 4; ++t) {
    int idx = wave*4 + t, mt = idx >> 4, rest = idx & 15, h = rest >> 1, et = rest & 1;
    bf16x16 af = frag_kmaj(sa + mt*16*264 + h*HD_, 264);
    bf16x16 bfv = frag_attB(attT + ((size_t)(bidx*H_ + h)*HD_ + et*16)*HD_);
    f32x8 c = zero8();
    c = wmma_bf16(af, bfv, c);
    int n = lane & 15;
#pragma unroll
    for (int r2 = 0; r2 < 8; ++r2)
      sy[(mt*16 + r2 + half*8)*256 + h*HD_ + et*16 + n] = c[r2];
  }
  styl_block(sy, sa, swt0, swt1, filmS, bidx, sg, sb, Wo, bo, xin, out, rowbase);
}

// ============ Kernel 8: FFN (W1/gelu/W2) + styl, fully fused per tile ============
__global__ void ffn_kernel(const float* xin,
                           const float* W1, const float* b1,
                           const float* W2, const float* b2,
                           const float* filmS, const float* sg, const float* sb,
                           const float* Wo, const float* bo, float* out) {
  extern __shared__ char smem[];
  float* sy   = (float*)smem;                     // [32][256]
  bf16*  sa   = (bf16*)(smem + 32768);            // [32][264]
  bf16*  swt0 = (bf16*)(smem + 49664);            // [256][40]
  bf16*  swt1 = (bf16*)(smem + 70144);            // [256][40]
  bf16*  su   = (bf16*)(smem + 90624);            // [32][1040]
  const int LDU = 1040;
  int tid = threadIdx.x, lane = tid & 31, wave = tid >> 5, half = lane >> 4, nn = lane & 15;
  long rowbase = (long)blockIdx.x * 32;
  int bidx = (int)(rowbase / T_);
  async_copy_tile_f32(xin + rowbase*D_, sy);
  __syncthreads();
#pragma unroll
  for (int u = 0; u < 8; ++u) {
    int idx = u*256 + tid;
    f32x4 v = ((const f32x4*)sy)[idx];
    int r = idx >> 6, c4 = idx & 63;
    bf16x4 pk;
#pragma unroll
    for (int i = 0; i < 4; ++i) pk[i] = (bf16)v[i];
    *(bf16x4*)(sa + r*264 + c4*4) = pk;
  }
  __syncthreads();
  // GEMM1: u = x @ W1 + b1, gelu -> su (bf16), 4 chunks of 256 cols
  for (int ch = 0; ch < 4; ++ch) {
    f32x8 acc[4];
#pragma unroll
    for (int t = 0; t < 4; ++t) acc[t] = zero8();
    stage_w_t(W1 + ch*256, FD_, swt0);
    __syncthreads();
    for (int kk = 0; kk < 8; ++kk) {
      bf16* cur = (kk & 1) ? swt1 : swt0;
      bf16* nxt = (kk & 1) ? swt0 : swt1;
      if (kk < 6) __builtin_prefetch(W1 + (size_t)(kk+2)*32*FD_ + ch*256, 0, 1);
      if (kk < 7) stage_w_t(W1 + (size_t)(kk+1)*32*FD_ + ch*256, FD_, nxt);
#pragma unroll
      for (int t = 0; t < 4; ++t) {
        int tile = wave*4 + t, mt = tile >> 4, nt = tile & 15;
        acc[t] = wmma_bf16(frag_kmaj(sa + mt*16*264 + kk*32, 264),
                           frag_kmaj(cur + nt*16*40, 40), acc[t]);
      }
      __syncthreads();
    }
#pragma unroll
    for (int t = 0; t < 4; ++t) {
      int tile = wave*4 + t, mt = tile >> 4, nt = tile & 15;
#pragma unroll
      for (int r2 = 0; r2 < 8; ++r2) {
        int rr = mt*16 + r2 + half*8;
        int cc = ch*256 + nt*16 + nn;
        su[rr*LDU + cc] = (bf16)gelu_f(acc[t][r2] + b1[cc]);
      }
    }
  }
  __syncthreads();
  // GEMM2: y = gelu(u) @ W2 + b2, K=1024 in 32 steps
  f32x8 acc2[4];
#pragma unroll
  for (int t = 0; t < 4; ++t) acc2[t] = zero8();
  stage_w_t(W2, D_, swt0);
  __syncthreads();
  for (int kk = 0; kk < 32; ++kk) {
    bf16* cur = (kk & 1) ? swt1 : swt0;
    bf16* nxt = (kk & 1) ? swt0 : swt1;
    if (kk < 30) __builtin_prefetch(W2 + (size_t)(kk+2)*32*D_, 0, 1);
    if (kk < 31) stage_w_t(W2 + (size_t)(kk+1)*32*D_, D_, nxt);
#pragma unroll
    for (int t = 0; t < 4; ++t) {
      int tile = wave*4 + t, mt = tile >> 4, nt = tile & 15;
      acc2[t] = wmma_bf16(frag_kmaj(su + mt*16*LDU + kk*32, LDU),
                          frag_kmaj(cur + nt*16*40, 40), acc2[t]);
    }
    __syncthreads();
  }
#pragma unroll
  for (int t = 0; t < 4; ++t) {
    int tile = wave*4 + t, mt = tile >> 4, nt = tile & 15;
#pragma unroll
    for (int r2 = 0; r2 < 8; ++r2)
      sy[(mt*16 + r2 + half*8)*256 + nt*16 + nn] = acc2[t][r2] + b2[nt*16 + nn];
  }
  styl_block(sy, sa, swt0, swt1, filmS, bidx, sg, sb, Wo, bo, xin, out, rowbase);
}

extern "C" void kernel_launch(void* const* d_in, const int* in_sizes, int n_in,
                              void* d_out, int out_size, void* d_ws, size_t ws_size,
                              hipStream_t stream) {
  (void)in_sizes; (void)n_in; (void)out_size; (void)ws_size;
  const float* x    = (const float*)d_in[0];
  const float* xf   = (const float*)d_in[1];
  const float* emb  = (const float*)d_in[2];
  const int*   mask = (const int*)d_in[3];
  const float* sa_ln_g = (const float*)d_in[4];
  const float* sa_ln_b = (const float*)d_in[5];
  const float* sa_Wq = (const float*)d_in[6];  const float* sa_bq = (const float*)d_in[7];
  const float* sa_Wk = (const float*)d_in[8];  const float* sa_bk = (const float*)d_in[9];
  const float* sa_Wv = (const float*)d_in[10]; const float* sa_bv = (const float*)d_in[11];
  const float* sa_We = (const float*)d_in[12]; const float* sa_be = (const float*)d_in[13];
  const float* sa_sg = (const float*)d_in[14]; const float* sa_sb = (const float*)d_in[15];
  const float* sa_Wo = (const float*)d_in[16]; const float* sa_bo = (const float*)d_in[17];
  const float* ca_ln_g = (const float*)d_in[18];
  const float* ca_ln_b = (const float*)d_in[19];
  const float* ca_tg = (const float*)d_in[20]; const float* ca_tb = (const float*)d_in[21];
  const float* ca_Wq = (const float*)d_in[22]; const float* ca_bq = (const float*)d_in[23];
  const float* ca_Wk = (const float*)d_in[24]; const float* ca_bk = (const float*)d_in[25];
  const float* ca_Wv = (const float*)d_in[26]; const float* ca_bv = (const float*)d_in[27];
  const float* ca_We = (const float*)d_in[28]; const float* ca_be = (const float*)d_in[29];
  const float* ca_sg = (const float*)d_in[30]; const float* ca_sb = (const float*)d_in[31];
  const float* ca_Wo = (const float*)d_in[32]; const float* ca_bo = (const float*)d_in[33];
  const float* f_W1 = (const float*)d_in[34]; const float* f_b1 = (const float*)d_in[35];
  const float* f_W2 = (const float*)d_in[36]; const float* f_b2 = (const float*)d_in[37];
  const float* f_We = (const float*)d_in[38]; const float* f_be = (const float*)d_in[39];
  const float* f_sg = (const float*)d_in[40]; const float* f_sb = (const float*)d_in[41];
  const float* f_Wo = (const float*)d_in[42]; const float* f_bo = (const float*)d_in[43];

  char* w = (char*)d_ws;
  bf16* qb = (bf16*)w; w += (size_t)BT_*D_*sizeof(bf16);
  bf16* kT = (bf16*)w; w += (size_t)BT_*D_*sizeof(bf16);
  bf16* vT = (bf16*)w; w += (size_t)BT_*D_*sizeof(bf16);
  float* colmax = (float*)w; w += (size_t)B_*D_*4;
  float* colinv = (float*)w; w += (size_t)B_*D_*4;
  float* att_sa = (float*)w; w += (size_t)B_*H_*HD_*HD_*4;
  float* att_ca = (float*)w; w += (size_t)B_*H_*HD_*HD_*4;
  float* film   = (float*)w; w += (size_t)3*B_*512*4;
  float* xout = (float*)d_out;

  const size_t LDS3 = 32768 + 16896 + 2*20480 + 32768;   // 123,392 B
  const size_t LDS6 = 32768 + 16896 + 2*20480;           //  90,624 B
  const size_t LDS8 = 32768 + 16896 + 2*20480 + 66560;   // 157,184 B
  hipFuncSetAttribute((const void*)sa_qkv_kernel, hipFuncAttributeMaxDynamicSharedMemorySize, (int)LDS3);
  hipFuncSetAttribute((const void*)sa_apply_kernel, hipFuncAttributeMaxDynamicSharedMemorySize, (int)LDS6);
  hipFuncSetAttribute((const void*)ca_apply_kernel, hipFuncAttributeMaxDynamicSharedMemorySize, (int)LDS6);
  hipFuncSetAttribute((const void*)ffn_kernel, hipFuncAttributeMaxDynamicSharedMemorySize, (int)LDS8);

  film_kernel<<<B_, 256, 0, stream>>>(emb, sa_We, sa_be, ca_We, ca_be, f_We, f_be, film);
  ca_ctx_kernel<<<B_*H_, 256, 0, stream>>>(xf, ca_tg, ca_tb, ca_Wk, ca_bk, ca_Wv, ca_bv, att_ca);
  sa_qkv_kernel<<<BT_/32, 256, LDS3, stream>>>(x, mask, sa_ln_g, sa_ln_b,
                                               sa_Wq, sa_bq, sa_Wk, sa_bk, sa_Wv, sa_bv,
                                               qb, kT, vT);
  sa_colstat_kernel<<<B_*D_, 256, 0, stream>>>(kT, colmax, colinv);
  sa_att_kernel<<<B_*H_, 256, 0, stream>>>(kT, vT, colmax, colinv, att_sa);
  sa_apply_kernel<<<BT_/32, 256, LDS6, stream>>>(qb, att_sa, film, sa_sg, sa_sb,
                                                 sa_Wo, sa_bo, x, xout);
  ca_apply_kernel<<<BT_/32, 256, LDS6, stream>>>(xout, ca_ln_g, ca_ln_b, ca_Wq, ca_bq,
                                                 att_ca, film + (size_t)B_*512,
                                                 ca_sg, ca_sb, ca_Wo, ca_bo, xout);
  ffn_kernel<<<BT_/32, 256, LDS8, stream>>>(xout, f_W1, f_b1, f_W2, f_b2,
                                            film + (size_t)2*B_*512,
                                            f_sg, f_sb, f_Wo, f_bo, xout);
}
